// EfficientAttention_71176198029846
// MI455X (gfx1250) — matrix operand order repaired and checked by reading, not compiled, not run on hardware
//
#include <hip/hip_runtime.h>

// ---------------------------------------------------------------------------
// EfficientAttention (linear attention) for MI455X / gfx1250, wave32 + WMMA.
//   B=4, L=4096, D=1024, H=16, dk=64
// All GEMMs on v_wmma_f32_16x16x32_f16 (f16 in, f32 accum), softmaxes in f32.
// GEMM: 256x64 block tile, 32x64 wave tile (8 WMMA/stage/wave), double-
// buffered LDS staged via global_load_async_to_lds_b128 (ASYNCcnt), one
// barrier per K-step.  Q-softmax fused in registers via __shfl_xor.
// Workspace (~143 MiB): xh | wq/wk/wv/wo h | qh | kh | vh | kvt ; oh aliases xh.
// ---------------------------------------------------------------------------

#define BATCH 4
#define SEQ 4096
#define DM 1024
#define NHEADS 16

#define USE_ASYNC_LDS 1   // flip to 0 if the async-to-LDS asm fails to assemble

typedef __attribute__((ext_vector_type(16))) _Float16 v16h;
typedef __attribute__((ext_vector_type(8)))  _Float16 v8h;
typedef __attribute__((ext_vector_type(8)))  float    v8f;

// 16B global -> LDS copy. Async path: GLOBAL_LOAD_ASYNC_TO_LDS_B128 (ISA
// §10.1/15.18, tracked by ASYNCcnt). LDS byte offset = low 32 bits of the
// generic pointer (flat-aperture rule, wave-relative).
static __device__ __forceinline__ void copy16B_to_lds(const _Float16* g,
                                                      _Float16* l) {
#if USE_ASYNC_LDS
  unsigned int loff = (unsigned int)(unsigned long long)l;
  unsigned long long gaddr = (unsigned long long)g;
  asm volatile("global_load_async_to_lds_b128 %0, %1, off"
               :: "v"(loff), "v"(gaddr) : "memory");
#else
  *(v8h*)l = *(const v8h*)g;
#endif
}

static __device__ __forceinline__ void wait_async_all() {
#if USE_ASYNC_LDS
#if __has_builtin(__builtin_amdgcn_s_wait_asynccnt)
  __builtin_amdgcn_s_wait_asynccnt(0);
#else
  asm volatile("s_wait_asynccnt 0x0" ::: "memory");
#endif
#endif
}

// A-fragment (16-bit A 16x32, this code's contiguous k order, not the ISA table's): lane l holds row m0+(l&15);
// halves h: k = kbase + h, kbase = (l>=16)*16.
static __device__ __forceinline__ v16h load_a_frag(const _Float16* tile, int m0) {
  int lane = threadIdx.x & 31;
  int kb = (lane >= 16) ? 16 : 0;
  const _Float16* p = tile + (m0 + (lane & 15)) * 32 + kb;
  v8h lo = *(const v8h*)p;
  v8h hi = *(const v8h*)(p + 8);
  return __builtin_shufflevector(lo, hi, 0,1,2,3,4,5,6,7,8,9,10,11,12,13,14,15);
}

// B-fragment (16-bit B 32x16): lane l holds column n0+(l&15);
// halves h: k = h + (l>=16 ? 16 : 0) -> one contiguous 32B chunk.
static __device__ __forceinline__ v16h load_b_frag(const _Float16* tile, int n0) {
  int lane = threadIdx.x & 31;
  int koff = (lane >= 16) ? 16 : 0;
  return *(const v16h*)(tile + (n0 + (lane & 15)) * 32 + koff);
}

// ---------------------------------------------------------------------------
// Generic tiled GEMM:  C[m][n] = sum_k A[m][k] * W[n][k]  (+ bias[n])
// Block: 256 threads / 8 waves; tile 256(M) x 64(N); K-step 32.
// Wave tile 32x64: 2 A-frags x 4 B-frags = 8 WMMAs per stage.
// MODE 0: f16 out.  MODE 1: f16 out + row-softmax over the 64-wide N tile
// (in-register, __shfl_xor over the 16-lane half-wave).  MODE 2: f32 out.
// Grouped mode via grid.z: bz -> (bb = bz/hh, hq = bz%hh) offsets.
// ---------------------------------------------------------------------------
template <int MODE>
__global__ __launch_bounds__(256)
void gemm_wmma_kernel(const _Float16* __restrict__ A, long lda,
                      const _Float16* __restrict__ W, long ldw,
                      const float* __restrict__ bias,
                      void* __restrict__ Cv, long ldc, int Kdim,
                      long sAb, long sAh, long sWb, long sCb, long sCh, int hh)
{
  __shared__ alignas(32) _Float16 Abuf[2][256 * 32];  // [buf][m][k]  2x16KB
  __shared__ alignas(32) _Float16 Bbuf[2][64 * 32];   // [buf][n][k]  2x4KB

  const int bz = blockIdx.z;
  const int bb = bz / hh, hq = bz % hh;
  A += (long)bb * sAb + (long)hq * sAh;
  W += (long)bz * sWb;
  const long coff = (long)bb * sCb + (long)hq * sCh;

  const int t = threadIdx.x;
  const int w = t >> 5;
  const int lane = t & 31;
  const int mbase = blockIdx.x * 256;
  const int nbase = blockIdx.y * 64;

  // staging decomposition: A tile 256x32h = 1024 16B-chunks (4/thread),
  // B tile 64x32h = 256 16B-chunks (1/thread)
  const int ar0 = t >> 2, ak0 = (t & 3) * 8;

  auto stage = [&](int buf, int k0) {
#pragma unroll
    for (int i = 0; i < 4; ++i) {
      int row = ar0 + i * 64;
      copy16B_to_lds(&A[(long)(mbase + row) * lda + k0 + ak0],
                     &Abuf[buf][row * 32 + ak0]);
    }
    copy16B_to_lds(&W[(long)(nbase + ar0) * ldw + k0 + ak0],
                   &Bbuf[buf][ar0 * 32 + ak0]);
  };

  v8f acc0[4] = {v8f{}, v8f{}, v8f{}, v8f{}};
  v8f acc1[4] = {v8f{}, v8f{}, v8f{}, v8f{}};

  const int nsteps = Kdim >> 5;
  stage(0, 0);
  for (int s = 0; s < nsteps; ++s) {
    wait_async_all();
    __syncthreads();  // buf[s&1] fully landed; everyone done with buf[(s+1)&1]
    if (s + 1 < nsteps) stage((s + 1) & 1, (s + 1) * 32);

    const _Float16* At = Abuf[s & 1];
    const _Float16* Bt = Bbuf[s & 1];
    v16h a0 = load_a_frag(At, w * 32);
    v16h a1 = load_a_frag(At, w * 32 + 16);
#pragma unroll
    for (int tt = 0; tt < 4; ++tt) {
      v16h bf = load_b_frag(Bt, tt * 16);
      acc0[tt] = __builtin_amdgcn_wmma_f32_16x16x32_f16(
          false, a0, false, bf, (short)0, acc0[tt], false, false);
      acc1[tt] = __builtin_amdgcn_wmma_f32_16x16x32_f16(
          false, a1, false, bf, (short)0, acc1[tt], false, false);
    }
  }

  // Epilogue. C/D layout: acc[tt][r] = C[m0 + r + 8*(lane>=16)][tt*16+(lane&15)]
  const int colc = lane & 15;
  const int halfsel = lane >> 4;
  float badd[4];
#pragma unroll
  for (int tt = 0; tt < 4; ++tt)
    badd[tt] = bias ? bias[nbase + tt * 16 + colc] : 0.0f;

#pragma unroll
  for (int aw = 0; aw < 2; ++aw) {
    v8f* acc = aw ? acc1 : acc0;
#pragma unroll
    for (int r = 0; r < 8; ++r) {
      const int m = mbase + w * 32 + aw * 16 + r + halfsel * 8;
      if (MODE == 1) {
        // row softmax over 64 columns: 4 accs x 16 lanes of this half-wave.
        float mx = -3.0e38f;
#pragma unroll
        for (int tt = 0; tt < 4; ++tt) {
          float v = acc[tt][r] + badd[tt];
          acc[tt][r] = v;
          mx = fmaxf(mx, v);
        }
        mx = fmaxf(mx, __shfl_xor(mx, 1));
        mx = fmaxf(mx, __shfl_xor(mx, 2));
        mx = fmaxf(mx, __shfl_xor(mx, 4));
        mx = fmaxf(mx, __shfl_xor(mx, 8));
        float sum = 0.0f;
#pragma unroll
        for (int tt = 0; tt < 4; ++tt) {
          float e = __expf(acc[tt][r] - mx);
          acc[tt][r] = e;
          sum += e;
        }
        sum += __shfl_xor(sum, 1);
        sum += __shfl_xor(sum, 2);
        sum += __shfl_xor(sum, 4);
        sum += __shfl_xor(sum, 8);
        float inv = 1.0f / sum;
        _Float16* C = (_Float16*)Cv + coff;
#pragma unroll
        for (int tt = 0; tt < 4; ++tt)
          C[(long)m * ldc + nbase + tt * 16 + colc] =
              (_Float16)(acc[tt][r] * inv);
      } else if (MODE == 0) {
        _Float16* C = (_Float16*)Cv + coff;
#pragma unroll
        for (int tt = 0; tt < 4; ++tt)
          C[(long)m * ldc + nbase + tt * 16 + colc] =
              (_Float16)(acc[tt][r] + badd[tt]);
      } else {
        float* C = (float*)Cv + coff;
#pragma unroll
        for (int tt = 0; tt < 4; ++tt)
          C[(long)m * ldc + nbase + tt * 16 + colc] = acc[tt][r] + badd[tt];
      }
    }
  }
}

// ---------------------------------------------------------------------------
// Column softmax for K over L (per (b, e) column), then multiply by mask[b,l].
// In-place on f16 K. One block per (b, 64-column group); f32 math.
// ---------------------------------------------------------------------------
__global__ __launch_bounds__(256)
void k_colsoftmax_kernel(_Float16* __restrict__ Kh, const int* __restrict__ mask)
{
  __shared__ float red[4][64];
  const int b = blockIdx.x >> 4;
  const int e0 = (blockIdx.x & 15) * 64;
  const int t = threadIdx.x;
  const int c = t & 63, g = t >> 6;
  const long base = (long)b * SEQ * DM + e0 + c;

  float mx = -3.0e38f;
  for (int l = g; l < SEQ; l += 4)
    mx = fmaxf(mx, (float)Kh[base + (long)l * DM]);
  red[g][c] = mx;
  __syncthreads();
  float m4 = fmaxf(fmaxf(red[0][c], red[1][c]), fmaxf(red[2][c], red[3][c]));
  __syncthreads();

  float s = 0.0f;
  for (int l = g; l < SEQ; l += 4)
    s += __expf((float)Kh[base + (long)l * DM] - m4);
  red[g][c] = s;
  __syncthreads();
  float inv = 1.0f / (red[0][c] + red[1][c] + red[2][c] + red[3][c]);

  for (int l = g; l < SEQ; l += 4) {
    float v = __expf((float)Kh[base + (long)l * DM] - m4) * inv *
              (float)mask[b * SEQ + l];
    Kh[base + (long)l * DM] = (_Float16)v;
  }
}

// ---------------------------------------------------------------------------
// KV[d][e] = sum_l K[b,l,h*64+d] * V[b,l,h*64+e]   (64x64xL per (b,h))
// One block per (b,h). 8 waves = 4 M-tiles x 2 K-slices, LDS-transposed
// K/V tiles (A needs K^T), f32 LDS reduction, stores KV^T for the next GEMM.
// ---------------------------------------------------------------------------
__global__ __launch_bounds__(256)
void kv_wmma_kernel(const _Float16* __restrict__ Kh,
                    const _Float16* __restrict__ Vh,
                    _Float16* __restrict__ KVt)
{
  __shared__ alignas(32) _Float16 Kt[2][64][32];  // [buf][d][l_local]
  __shared__ alignas(32) _Float16 Vt[2][64][32];  // [buf][e][l_local]
  __shared__ float Cred[8][16][64];

  const int bz = blockIdx.x;
  const int b = bz >> 4, h = bz & 15;
  const long base = (long)b * SEQ * DM + h * 64;

  const int t = threadIdx.x, w = t >> 5, lane = t & 31;
  const int mw = w & 3, ks = w >> 2;

  v8f acc[4] = {v8f{}, v8f{}, v8f{}, v8f{}};

  const int ll = t >> 3;        // 0..31  (row within 32-step slice)
  const int d0 = (t & 7) * 8;   // 0..56  (8 contiguous channels)

  for (int dstep = 0; dstep < SEQ / 64; ++dstep) {
    const int l0 = dstep * 64;
#pragma unroll
    for (int q = 0; q < 2; ++q) {
      long g = base + (long)(l0 + q * 32 + ll) * DM + d0;
      v8h kv8 = *(const v8h*)&Kh[g];
      v8h vv8 = *(const v8h*)&Vh[g];
#pragma unroll
      for (int j = 0; j < 8; ++j) {  // transpose into LDS
        Kt[q][d0 + j][ll] = kv8[j];
        Vt[q][d0 + j][ll] = vv8[j];
      }
    }
    __syncthreads();

    v16h a = load_a_frag(&Kt[ks][0][0], mw * 16);
#pragma unroll
    for (int tt = 0; tt < 4; ++tt) {
      v16h bf = load_b_frag(&Vt[ks][0][0], tt * 16);
      acc[tt] = __builtin_amdgcn_wmma_f32_16x16x32_f16(
          false, a, false, bf, (short)0, acc[tt], false, false);
    }
    __syncthreads();
  }

  const int colc = lane & 15, halfsel = lane >> 4;
#pragma unroll
  for (int tt = 0; tt < 4; ++tt)
#pragma unroll
    for (int r = 0; r < 8; ++r)
      Cred[w][r + halfsel * 8][tt * 16 + colc] = acc[tt][r];
  __syncthreads();

  for (int i = t; i < 64 * 64; i += 256) {
    int m = i >> 6, n = i & 63;
    float s = Cred[m >> 4][m & 15][n] + Cred[4 + (m >> 4)][m & 15][n];
    KVt[(long)bz * 4096 + n * 64 + m] = (_Float16)s;  // store transposed
  }
}

// ---------------------------------------------------------------------------
__global__ __launch_bounds__(256)
void cvt_f32_f16_kernel(const float* __restrict__ s, _Float16* __restrict__ d,
                        long n)
{
  for (long i = (long)blockIdx.x * blockDim.x + threadIdx.x; i < n;
       i += (long)gridDim.x * blockDim.x)
    d[i] = (_Float16)s[i];
}

// ---------------------------------------------------------------------------
extern "C" void kernel_launch(void* const* d_in, const int* in_sizes, int n_in,
                              void* d_out, int out_size, void* d_ws,
                              size_t ws_size, hipStream_t stream)
{
  (void)in_sizes; (void)n_in; (void)out_size; (void)ws_size;
  const float* x  = (const float*)d_in[0];
  const int* mask = (const int*)d_in[1];
  const float* Wq = (const float*)d_in[2];
  const float* bq = (const float*)d_in[3];
  const float* Wk = (const float*)d_in[4];
  const float* bk = (const float*)d_in[5];
  const float* Wv = (const float*)d_in[6];
  const float* bv = (const float*)d_in[7];
  const float* Wo = (const float*)d_in[8];
  const float* bo = (const float*)d_in[9];
  float* out = (float*)d_out;

  const long NX = (long)BATCH * SEQ * DM;  // 16,777,216
  const long NW = (long)DM * DM;           //  1,048,576

  char* p = (char*)d_ws;
  _Float16* xh  = (_Float16*)p;  p += NX * 2;
  _Float16* wqh = (_Float16*)p;  p += NW * 2;
  _Float16* wkh = (_Float16*)p;  p += NW * 2;
  _Float16* wvh = (_Float16*)p;  p += NW * 2;
  _Float16* woh = (_Float16*)p;  p += NW * 2;
  _Float16* qh  = (_Float16*)p;  p += NX * 2;
  _Float16* kh  = (_Float16*)p;  p += NX * 2;
  _Float16* vh  = (_Float16*)p;  p += NX * 2;
  _Float16* kvt = (_Float16*)p;  p += (long)BATCH * NHEADS * 64 * 64 * 2;
  _Float16* oh  = xh;  // x is dead after the three projections

  dim3 blk(256);

  cvt_f32_f16_kernel<<<2048, blk, 0, stream>>>(x,  xh,  NX);
  cvt_f32_f16_kernel<<<512,  blk, 0, stream>>>(Wq, wqh, NW);
  cvt_f32_f16_kernel<<<512,  blk, 0, stream>>>(Wk, wkh, NW);
  cvt_f32_f16_kernel<<<512,  blk, 0, stream>>>(Wv, wvh, NW);
  cvt_f32_f16_kernel<<<512,  blk, 0, stream>>>(Wo, woh, NW);

  // Q/K/V projections: (16384 x 1024) x (1024 x 1024), 256-row block tiles
  dim3 gproj(64, 16, 1);
  gemm_wmma_kernel<1><<<gproj, blk, 0, stream>>>(  // Q + fused dk-softmax
      xh, DM, wqh, DM, bq, (void*)qh, DM, DM, 0, 0, 0, 0, 0, 1);
  gemm_wmma_kernel<0><<<gproj, blk, 0, stream>>>(  // K raw
      xh, DM, wkh, DM, bk, (void*)kh, DM, DM, 0, 0, 0, 0, 0, 1);
  gemm_wmma_kernel<0><<<gproj, blk, 0, stream>>>(  // V
      xh, DM, wvh, DM, bv, (void*)vh, DM, DM, 0, 0, 0, 0, 0, 1);

  // K: softmax over L per (b,h,d) channel, then * mask   (in place)
  k_colsoftmax_kernel<<<64, blk, 0, stream>>>(kh, mask);

  // KV = K^T V per (b,h)  -> stored transposed
  kv_wmma_kernel<<<64, blk, 0, stream>>>(kh, vh, kvt);

  // out = Q @ KV per (b,h): grouped GEMM, M=4096, N=64, K=64
  dim3 gapply(SEQ / 256, 1, BATCH * NHEADS);
  gemm_wmma_kernel<0><<<gapply, blk, 0, stream>>>(
      qh, DM, kvt, 64, nullptr, (void*)oh, DM, 64,
      (long)SEQ * DM, 64, 4096, (long)SEQ * DM, 64, NHEADS);

  // out projection: f32 output + bias -> d_out
  gemm_wmma_kernel<2><<<gproj, blk, 0, stream>>>(
      oh, DM, woh, DM, bo, (void*)out, DM, DM, 0, 0, 0, 0, 0, 1);
}
